// AttentionOperator_36129264894206
// MI455X (gfx1250) — matrix operand order repaired and checked by reading, not compiled, not run on hardware
//
#include <hip/hip_runtime.h>
#include <hip/hip_bf16.h>
#include <math.h>
#include <stdint.h>

// ---------------------------------------------------------------------------
// Problem constants (from the reference)
// ---------------------------------------------------------------------------
#define HID     256
#define TX      512
#define TY      1000
#define TY_PAD  1008            // TY rounded up to 16 for WMMA M tiles
#define NEGV    (-1000000000.0f)

typedef _Float16 v16h __attribute__((ext_vector_type(16)));
typedef _Float16 v8h  __attribute__((ext_vector_type(8)));
typedef float    v8f  __attribute__((ext_vector_type(8)));

// Generic (flat) shared-memory pointer -> 32-bit LDS byte offset, as required
// by the VDST operand of global_load_async_to_lds_*.
__device__ __forceinline__ uint32_t to_lds_off(const void* p) {
    return (uint32_t)(uint64_t)(__attribute__((address_space(3))) const void*)p;
}

// ---------------------------------------------------------------------------
// Sinusoid positional value: table[p][h] = sin/cos(p / 10000^(2*(h//2)/HID))
// Computed in double to match the numpy float64 table closely.
// ---------------------------------------------------------------------------
__device__ __forceinline__ float pos_val(int p, int h) {
    int j2 = h & ~1;
    double inv = exp(-(double)j2 * (9.210340371976182736 / (double)HID)); // ln(10000)
    double ang = (double)p * inv;
    return (float)((h & 1) ? cos(ang) : sin(ang));
}

// ---------------------------------------------------------------------------
// K1a: x_aug[b][s][h] = x_h[b][h][s] + pos[s][h], stored as f16 (col-major
// per s column so WMMA B fragments are contiguous 16B loads).
// ---------------------------------------------------------------------------
__global__ __launch_bounds__(256) void build_xaug(const float* __restrict__ xh,
                                                  _Float16* __restrict__ xaug, int B) {
    size_t idx = (size_t)blockIdx.x * blockDim.x + threadIdx.x;
    size_t total = (size_t)B * TX * HID;
    if (idx >= total) return;
    int h = (int)(idx % HID);
    int s = (int)((idx / HID) % TX);
    int b = (int)(idx / ((size_t)HID * TX));
    float v = xh[((size_t)b * HID + h) * TX + s] + pos_val(s, h);
    xaug[idx] = (_Float16)v;
}

// K1b: y_aug[b][t][h], zero-padded rows for t in [TY, TY_PAD)
__global__ __launch_bounds__(256) void build_yaug(const float* __restrict__ yh,
                                                  _Float16* __restrict__ yaug, int B) {
    size_t idx = (size_t)blockIdx.x * blockDim.x + threadIdx.x;
    size_t total = (size_t)B * TY_PAD * HID;
    if (idx >= total) return;
    int h = (int)(idx % HID);
    int t = (int)((idx / HID) % TY_PAD);
    int b = (int)(idx / ((size_t)HID * TY_PAD));
    float v = 0.0f;
    if (t < TY) v = yh[((size_t)b * HID + h) * TY + t] + pos_val(t, h);
    yaug[idx] = (_Float16)v;
}

// ---------------------------------------------------------------------------
// K2: fused scores GEMM (WMMA f16 -> f32) + masked softmax over s (512) +
//     pi_dummy[b,t] = sum_s alpha * (s * xm[s]).
//
// One workgroup = one 16-row t tile x all 512 s columns, 8 waves x 4 N-tiles.
// The shared 16x256 y tile (8 KB) is staged into LDS once via the CDNA5
// async copy path (global_load_async_to_lds_b128 / ASYNCcnt), then A
// fragments are served from LDS (ds_load_b128). B fragments stay on the
// direct global path: every byte is read exactly once per workgroup and the
// f16 working set is L2-resident.
// ---------------------------------------------------------------------------
__global__ __launch_bounds__(256) void scores_pidummy(
    const _Float16* __restrict__ xaug, const _Float16* __restrict__ yaug,
    const unsigned char* __restrict__ xmask,
    float* __restrict__ pdummy, int B)
{
    const int mtiles = (TY + 15) / 16;             // 63
    const int b  = blockIdx.x / mtiles;
    const int mt = blockIdx.x % mtiles;
    const int tid  = threadIdx.x;
    const int lane = tid & 31;
    const int wave = tid >> 5;
    const int lmod = lane & 15;
    const int half = lane >> 4;

    const _Float16* ytile = yaug + ((size_t)b * TY_PAD + (size_t)mt * 16) * HID;
    const _Float16* xb    = xaug + (size_t)b * TX * HID;
    const int n0 = wave * 64;

    __shared__ _Float16 shY[16 * HID];             // 8 KB shared A tile

    // ---- async stage of the y tile: 512 x 16B lane-transfers -------------
    {
        const uint32_t base = to_lds_off(shY);
        #pragma unroll
        for (int i = 0; i < 2; ++i) {
            uint32_t slot = (uint32_t)tid + (uint32_t)i * 256u;   // 0..511
            uint32_t ldsOff = base + slot * 16u;
            const char* g = (const char*)ytile + (size_t)slot * 16u;
            asm volatile("global_load_async_to_lds_b128 %0, %1, off"
                         :: "v"(ldsOff), "v"(g) : "memory");
        }
        asm volatile("s_wait_asynccnt 0x0" ::: "memory");
    }
    __syncthreads();

    // column masks / p values for this wave's 4 N tiles (per-lane column lmod)
    bool  msk[4];
    float pcol[4];
    #pragma unroll
    for (int j = 0; j < 4; ++j) {
        int s = n0 + j * 16 + lmod;
        msk[j]  = xmask[(size_t)b * TX + s] != 0;
        pcol[j] = msk[j] ? (float)s : 0.0f;
    }

    v8f acc[4] = {v8f{}, v8f{}, v8f{}, v8f{}};

    for (int kk = 0; kk < HID; kk += 32) {
        // A fragment (16x32 f16) from LDS: row M = lmod;
        // 16 contiguous K values ab..ab+15, ab = kk + (half?16:0)
        const _Float16* arow = shY + (size_t)lmod * HID + kk + (half ? 16 : 0);
        v8h a_lo = *(const v8h*)(arow);            // ds_load_b128
        v8h a_hi = *(const v8h*)(arow + 8);        // ds_load_b128
        v16h afrag = __builtin_shufflevector(a_lo, a_hi,
                        0,1,2,3,4,5,6,7,8,9,10,11,12,13,14,15);
        #pragma unroll
        for (int j = 0; j < 4; ++j) {
            // B fragment (32x16 f16): col N = lmod; 16 contiguous K values
            const _Float16* bcol = xb + (size_t)(n0 + j * 16 + lmod) * HID
                                      + kk + (half ? 16 : 0);
            v8h b_lo = *(const v8h*)(bcol);
            v8h b_hi = *(const v8h*)(bcol + 8);
            v16h bfrag = __builtin_shufflevector(b_lo, b_hi,
                            0,1,2,3,4,5,6,7,8,9,10,11,12,13,14,15);
            acc[j] = __builtin_amdgcn_wmma_f32_16x16x32_f16(
                         false, afrag, false, bfrag,
                         (short)0, acc[j], false, false);
        }
    }

    // ---- epilogue: scale, mask, stable softmax + weighted reduction -------
    __shared__ float redMax[8][16];
    __shared__ float redSE [8][16];
    __shared__ float redSP [8][16];

    const float scale = 0.0625f;   // 1/sqrt(256)
    float rm[8];
    #pragma unroll
    for (int i = 0; i < 8; ++i) {
        float m = -3.4e38f;
        #pragma unroll
        for (int j = 0; j < 4; ++j) {
            float v = acc[j][i] * scale;
            if (!msk[j]) v = NEGV;
            acc[j][i] = v;
            m = fmaxf(m, v);
        }
        // butterfly over the 16 lanes of this half (rows stay separated)
        #pragma unroll
        for (int ofs = 1; ofs < 16; ofs <<= 1)
            m = fmaxf(m, __shfl_xor(m, ofs));
        rm[i] = m;
    }
    if (lmod == 0) {
        #pragma unroll
        for (int i = 0; i < 8; ++i) redMax[wave][half * 8 + i] = rm[i];
    }
    __syncthreads();

    float grm[8];
    #pragma unroll
    for (int i = 0; i < 8; ++i) {
        float m = -3.4e38f;
        #pragma unroll
        for (int w = 0; w < 8; ++w)
            m = fmaxf(m, redMax[w][half * 8 + i]);
        grm[i] = m;
    }

    #pragma unroll
    for (int i = 0; i < 8; ++i) {
        float se = 0.0f, sp = 0.0f;
        #pragma unroll
        for (int j = 0; j < 4; ++j) {
            float e = __expf(acc[j][i] - grm[i]);
            se += e;
            sp += e * pcol[j];
        }
        #pragma unroll
        for (int ofs = 1; ofs < 16; ofs <<= 1) {
            se += __shfl_xor(se, ofs);
            sp += __shfl_xor(sp, ofs);
        }
        if (lmod == 0) {
            redSE[wave][half * 8 + i] = se;
            redSP[wave][half * 8 + i] = sp;
        }
    }
    __syncthreads();

    if (wave == 0 && lane < 16) {
        int t = mt * 16 + lane;
        if (t < TY) {
            float SE = 0.0f, SP = 0.0f;
            #pragma unroll
            for (int w = 0; w < 8; ++w) { SE += redSE[w][lane]; SP += redSP[w][lane]; }
            pdummy[(size_t)b * TY + t] = SP / SE;
        }
    }
}

// ---------------------------------------------------------------------------
// K3: per-batch serial scan: delta, fwd+bwd cumsum, normalization -> pi.
//     Also records mask popcounts for the final assembly.
// ---------------------------------------------------------------------------
__global__ void pi_scan(const float* __restrict__ pd,
                        const unsigned char* __restrict__ xm,
                        const unsigned char* __restrict__ ym,
                        float* __restrict__ delta, float* __restrict__ pi,
                        int* __restrict__ cnts, int B)
{
    int b = blockIdx.x;
    if (threadIdx.x != 0 || b >= B) return;
    const float* pdb = pd + (size_t)b * TY;
    const unsigned char* ymb = ym + (size_t)b * TY;
    const unsigned char* xmb = xm + (size_t)b * TX;
    float* db  = delta + (size_t)b * TY;
    float* pib = pi    + (size_t)b * TY;

    int xs = 0; for (int s = 0; s < TX; ++s) xs += xmb[s] ? 1 : 0;
    int ys = 0; for (int y = 0; y < TY; ++y) ys += ymb[y] ? 1 : 0;
    cnts[2 * b] = xs; cnts[2 * b + 1] = ys;

    float pf = 0.0f;
    for (int y = 0; y < TY; ++y) {
        float d = (y > 0) ? fmaxf(pdb[y] - pdb[y - 1], 0.0f) : 0.0f;
        d *= ymb[y] ? 1.0f : 0.0f;
        db[y] = d; pf += d; pib[y] = pf;          // cumsum
    }
    float S = pf;
    float mx = -3.4e38f;
    for (int y = 0; y < TY; ++y) {
        float ymv = ymb[y] ? 1.0f : 0.0f;
        float v = pib[y] * ymv - (S - pib[y] + db[y]); // pi_f + pi_b
        pib[y] = v;
        mx = fmaxf(mx, v);
    }
    float last  = fmaxf(mx, 1e-8f);
    float first = pib[0];
    float inv   = ((float)xs - 1.0f) / (last - first);
    for (int y = 0; y < TY; ++y) pib[y] = (pib[y] - first) * inv;
}

// ---------------------------------------------------------------------------
// K4: align(): one block per (b,t); wave 0 -> centers = p, wave 1 -> boundary.
//     Online softmax over Ty with shuffle merge.
// ---------------------------------------------------------------------------
__global__ __launch_bounds__(64) void align_kernel(
    const float* __restrict__ pi, const unsigned char* __restrict__ xm,
    const unsigned char* __restrict__ ym, const float* __restrict__ sigma_p,
    float* __restrict__ e_out, float* __restrict__ a_buf, int B)
{
    int b = blockIdx.x / TX;
    int t = blockIdx.x % TX;
    int lane = threadIdx.x & 31;
    int wave = threadIdx.x >> 5;
    float sigma = sigma_p[0];
    float xmv = xm[(size_t)b * TX + t] ? 1.0f : 0.0f;
    float p = (float)t * xmv;
    float c = (wave == 0) ? p : fmaxf(p - 0.5f, 0.0f);
    const float* pib = pi + (size_t)b * TY;
    const unsigned char* ymb = ym + (size_t)b * TY;

    float m = -3.4e38f, se = 0.0f, sq = 0.0f;
    for (int y = lane; y < TY; y += 32) {
        float d  = pib[y] - c;
        bool  mv = ymb[y] != 0;
        float en = mv ? (-(d * d) * sigma) : NEGV;
        float q  = mv ? (float)y : 0.0f;
        if (en > m) {
            float r = __expf(m - en);
            se = se * r + 1.0f;
            sq = sq * r + q;
            m = en;
        } else {
            float e = __expf(en - m);
            se += e; sq += e * q;
        }
    }
    #pragma unroll
    for (int ofs = 1; ofs < 32; ofs <<= 1) {
        float m2  = __shfl_xor(m,  ofs);
        float se2 = __shfl_xor(se, ofs);
        float sq2 = __shfl_xor(sq, ofs);
        float M = fmaxf(m, m2);
        float r1 = __expf(m - M), r2 = __expf(m2 - M);
        se = se * r1 + se2 * r2;
        sq = sq * r1 + sq2 * r2;
        m = M;
    }
    if (lane == 0) {
        float res = (sq / se) * xmv;
        if (wave == 0) e_out[(size_t)b * TX + t] = res;
        else           a_buf[(size_t)b * TX + t] = res;
    }
}

// ---------------------------------------------------------------------------
// K5: assemble a_real / b_real with the max_x -> max_y fixup.
// ---------------------------------------------------------------------------
__global__ __launch_bounds__(256) void assemble_kernel(
    const float* __restrict__ a_buf, const int* __restrict__ cnts,
    float* __restrict__ out_a, float* __restrict__ out_b, int B)
{
    int idx = blockIdx.x * blockDim.x + threadIdx.x;
    if (idx >= B * TX) return;
    int b = idx / TX, t = idx % TX;
    float ar = (t == 0) ? 0.0f : a_buf[idx];
    float bv = (t < TX - 1) ? a_buf[idx + 1] : 0.0f;  // a_real[t+1] == a_buf[t+1]
    if (t == cnts[2 * b] - 1) bv = (float)(cnts[2 * b + 1] - 1);
    out_a[idx] = ar;
    out_b[idx] = bv;
}

// ---------------------------------------------------------------------------
// Launch
// ---------------------------------------------------------------------------
extern "C" void kernel_launch(void* const* d_in, const int* in_sizes, int n_in,
                              void* d_out, int out_size, void* d_ws, size_t ws_size,
                              hipStream_t stream) {
    const int B = in_sizes[0] / (HID * TX);
    const float*         xh    = (const float*)d_in[0];
    const float*         yh    = (const float*)d_in[1];
    const unsigned char* xmask = (const unsigned char*)d_in[2];  // jnp bool = 1 byte
    const unsigned char* ymask = (const unsigned char*)d_in[3];
    const float*         sigma = (const float*)d_in[4];

    // workspace layout (256B aligned sections), ~26 MB total for B=32
    char* ws = (char*)d_ws;
    size_t off = 0;
    auto carve = [&](size_t bytes) {
        void* p = ws + off;
        off += (bytes + 255) & ~(size_t)255;
        return p;
    };
    _Float16* xaug   = (_Float16*)carve((size_t)B * TX     * HID * sizeof(_Float16));
    _Float16* yaug   = (_Float16*)carve((size_t)B * TY_PAD * HID * sizeof(_Float16));
    float*    pdummy = (float*)   carve((size_t)B * TY * sizeof(float));
    float*    delta  = (float*)   carve((size_t)B * TY * sizeof(float));
    float*    pi     = (float*)   carve((size_t)B * TY * sizeof(float));
    float*    a_buf  = (float*)   carve((size_t)B * TX * sizeof(float));
    int*      cnts   = (int*)     carve((size_t)2 * B * sizeof(int));

    float* e_out = (float*)d_out;
    float* a_out = e_out + (size_t)B * TX;
    float* b_out = a_out + (size_t)B * TX;

    {
        size_t total = (size_t)B * TX * HID;
        build_xaug<<<(unsigned)((total + 255) / 256), 256, 0, stream>>>(xh, xaug, B);
    }
    {
        size_t total = (size_t)B * TY_PAD * HID;
        build_yaug<<<(unsigned)((total + 255) / 256), 256, 0, stream>>>(yh, yaug, B);
    }
    {
        const int mtiles = (TY + 15) / 16;  // 63
        scores_pidummy<<<B * mtiles, 256, 0, stream>>>(xaug, yaug, xmask, pdummy, B);
    }
    pi_scan<<<B, 32, 0, stream>>>(pdummy, xmask, ymask, delta, pi, cnts, B);
    align_kernel<<<B * TX, 64, 0, stream>>>(pi, xmask, ymask, sigma, e_out, a_buf, B);
    assemble_kernel<<<(B * TX + 255) / 256, 256, 0, stream>>>(a_buf, cnts, a_out, b_out, B);
}
